// SelfAttention_31379031065087
// MI455X (gfx1250) — matrix operand order repaired and checked by reading, not compiled, not run on hardware
//
#include <hip/hip_runtime.h>
#include <hip/hip_bf16.h>
#include <math.h>

typedef __bf16 bf16_t;
typedef __attribute__((ext_vector_type(16))) __bf16 v16bf;
typedef __attribute__((ext_vector_type(8)))  __bf16 v8bf;
typedef __attribute__((ext_vector_type(4)))  __bf16 v4bf;
typedef __attribute__((ext_vector_type(8)))  float  v8f;

#define DEVINL __device__ __forceinline__

constexpr int Bsz = 2;
constexpr int Nlen = 2048;
constexpr int Cdim = 1024;
constexpr int Hn = 16;
constexpr int HD = 64;
constexpr int TC = 3 * Cdim;    // 3072
constexpr int Mrows = Bsz * Nlen; // 4096

// ---------------- WMMA fragment helpers (layouts per CDNA5 ISA) ---
// A 16x32 bf16: lane = (row 0..15) + 16*(K-half); per lane two contiguous
// 16-byte chunks at K = half*8 and K = 16 + half*8.
DEVINL v16bf load_frag_a(const bf16_t* base, int ld) {
  const int lane = threadIdx.x & 31;
  const int row = lane & 15, half = lane >> 4;
  union { v16bf v; v8bf h[2]; } u;
  const bf16_t* p = base + row * ld + half * 8;
  u.h[0] = *(const v8bf*)(p);
  u.h[1] = *(const v8bf*)(p + 16);
  return u.v;
}

// B 32x16 bf16 (K-major operand rows): lane = (col 0..15) + 16*(K-half);
// per lane two contiguous 16-byte chunks at K = half*8 and K = 16 + half*8.
DEVINL v16bf load_frag_b(const bf16_t* base, int ld) {
  const int lane = threadIdx.x & 31;
  const int col = lane & 15, half = lane >> 4;
  union { v16bf v; v8bf h[2]; } u;
  const bf16_t* p = base + col * ld + half * 8;
  u.h[0] = *(const v8bf*)(p);
  u.h[1] = *(const v8bf*)(p + 16);
  return u.v;
}

DEVINL v8f wmma_bf16(v16bf a, v16bf b, v8f c) {
  return __builtin_amdgcn_wmma_f32_16x16x32_bf16(
      /*neg_a=*/false, a, /*neg_b=*/false, b,
      /*c_mod=*/(short)0, c, /*reuse_a=*/false, /*reuse_b=*/false);
}

// Per-lane async copy of 16 bytes global -> LDS, tracked by ASYNCcnt.
// (CDNA5 ISA GLOBAL_LOAD_ASYNC_TO_LDS_B128, GV mode)
DEVINL void async_copy_b128(unsigned lds_addr, const void* gaddr) {
  asm volatile("global_load_async_to_lds_b128 %0, %1, off"
               :: "v"(lds_addr), "v"(gaddr) : "memory");
}

// ---------------- fp32 -> bf16 conversion (x4 vectorized) ----------------
__global__ void __launch_bounds__(256)
cvt_f32_bf16(const float* __restrict__ in, bf16_t* __restrict__ out, int n4) {
  int i = blockIdx.x * blockDim.x + threadIdx.x;
  if (i < n4) {
    float4 v = ((const float4*)in)[i];
    v4bf o = { (__bf16)v.x, (__bf16)v.y, (__bf16)v.z, (__bf16)v.w };
    ((v4bf*)out)[i] = o;
  }
}

// ---------------- QKV GEMM + fused RoPE epilogue ----------------
// out[m, d] = sum_c X[m,c] * W[d,c] ; one wave per 64x64 tile.
// Tile N-range lies entirely inside one head of one of {Q,K,V}.
__global__ void __launch_bounds__(32)
qkv_rope_kernel(const bf16_t* __restrict__ X, const bf16_t* __restrict__ W,
                const float* __restrict__ sinp, const float* __restrict__ cosp,
                bf16_t* __restrict__ Qb, bf16_t* __restrict__ Kb,
                bf16_t* __restrict__ Vt) {
  const int m0 = blockIdx.x * 64;
  const int n0 = blockIdx.y * 64;
  const int lane = threadIdx.x & 31;
  const int col = lane & 15, half = lane >> 4;

  v8f acc[4][4];
#pragma unroll
  for (int i = 0; i < 4; ++i)
#pragma unroll
    for (int j = 0; j < 4; ++j) { v8f z = {}; acc[i][j] = z; }

  for (int k0 = 0; k0 < Cdim; k0 += 32) {
    if (k0 + 32 < Cdim) {  // hint next k-slab into cache (global_prefetch_b8)
      __builtin_prefetch(X + (m0 + 2 * lane) * Cdim + k0 + 32, 0, 1);
      __builtin_prefetch(W + (n0 + 2 * lane) * Cdim + k0 + 32, 0, 1);
    }
    v16bf a[4], b[4];
#pragma unroll
    for (int i = 0; i < 4; ++i) a[i] = load_frag_a(X + (m0 + 16 * i) * Cdim + k0, Cdim);
#pragma unroll
    for (int j = 0; j < 4; ++j) b[j] = load_frag_b(W + (n0 + 16 * j) * Cdim + k0, Cdim);
#pragma unroll
    for (int i = 0; i < 4; ++i)
#pragma unroll
      for (int j = 0; j < 4; ++j)
        acc[i][j] = wmma_bf16(a[i], b[j], acc[i][j]);
  }

  const int region = n0 / Cdim;        // 0=Q, 1=K, 2=V
  const int h = (n0 % Cdim) / HD;
  const int bidx = m0 / Nlen;
  const int t0 = m0 % Nlen;

  if (region == 2) {
    // V stored transposed: Vt[b,h,d,t] so P@V B-fragments are contiguous.
    bf16_t* base = Vt + (size_t)(bidx * Hn + h) * HD * Nlen;
#pragma unroll
    for (int i = 0; i < 4; ++i)
#pragma unroll
      for (int sub = 0; sub < 4; ++sub)
#pragma unroll
        for (int r = 0; r < 8; ++r) {
          int t = t0 + 16 * i + r + 8 * half;
          int d = sub * 16 + col;
          base[(size_t)d * Nlen + t] = (bf16_t)acc[i][sub][r];
        }
  } else {
    // Q or K: apply RoPE in-register. Pair columns (d, d+32) live in
    // accumulator sub-tiles (sub, sub+2) on the same lane / same VGPR slot.
    bf16_t* dst = (region == 0 ? Qb : Kb) + (size_t)(bidx * Hn + h) * Nlen * HD;
#pragma unroll
    for (int i = 0; i < 4; ++i)
#pragma unroll
      for (int sub = 0; sub < 2; ++sub)
#pragma unroll
        for (int r = 0; r < 8; ++r) {
          int t = t0 + 16 * i + r + 8 * half;
          int d1 = sub * 16 + col, d2 = d1 + 32;
          float v1 = acc[i][sub][r], v2 = acc[i][sub + 2][r];
          float o1 = v1, o2 = v2;
          if (t >= 1) {  // prefix token 0 left unrotated
            const float* sp = sinp + (size_t)(t - 1) * HD;
            const float* cp = cosp + (size_t)(t - 1) * HD;
            o1 = v1 * cp[d1] - v2 * sp[d1];
            o2 = v2 * cp[d2] + v1 * sp[d2];
          }
          bf16_t* row = dst + (size_t)t * HD;
          row[d1] = (bf16_t)o1;
          row[d2] = (bf16_t)o2;
        }
  }
}

// ---------------- Flash attention: one (b,h) x 64-query tile per block ------
// 4 waves; each wave owns 16 query rows. K/V 32-key tiles are staged into
// LDS with double-buffered ASYNC copies (ASYNCcnt) shared by all 4 waves.
__global__ void __launch_bounds__(128)
attn_kernel(const bf16_t* __restrict__ Qb, const bf16_t* __restrict__ Kb,
            const bf16_t* __restrict__ Vt, bf16_t* __restrict__ Ob) {
  const int bh = blockIdx.x;            // 0..31
  const int h = bh & 15, bidx = bh >> 4;
  const int w = threadIdx.x >> 5;
  const int lane = threadIdx.x & 31;
  const int col = lane & 15, half = lane >> 4;
  const int tid = threadIdx.x;
  const int m0 = blockIdx.y * 64 + w * 16;
  const float scale = 0.125f;           // hd^-0.5

  __shared__ bf16_t Ks[2][32][64];      // [buf][key][d]     8 KB
  __shared__ bf16_t Vs[2][64][32];      // [buf][d][key]     8 KB
  __shared__ bf16_t Pb[4][16][32];      // per-wave P stage   4 KB

  const bf16_t* Qp = Qb + ((size_t)bh * Nlen + m0) * HD;
  v16bf aq0 = load_frag_a(Qp, HD);        // K-dim (hd) 0..31
  v16bf aq1 = load_frag_a(Qp + 32, HD);   // K-dim (hd) 32..63

  float mrow[8], lrow[8];
#pragma unroll
  for (int r = 0; r < 8; ++r) { mrow[r] = -3.0e38f; lrow[r] = 0.0f; }
  v8f o[4];
#pragma unroll
  for (int n = 0; n < 4; ++n) { v8f z = {}; o[n] = z; }

  const char* Kg = (const char*)(Kb + (size_t)bh * Nlen * HD);
  const char* Vg = (const char*)(Vt + (size_t)bh * HD * Nlen);

  // Stage one 32-key tile: K tile is 4KB contiguous; V tile is 64 rows x 64B
  // (Vt row stride 4096B). 4 async b128 per thread => 4 ASYNCcnt per wave.
  auto stage_tile = [&](int kb, int buf) {
    const char* ksrc = Kg + (size_t)kb * (HD * 2);
    unsigned kdst = (unsigned)(uintptr_t)&Ks[buf][0][0];
#pragma unroll
    for (int c = 0; c < 2; ++c) {
      int j = tid + 128 * c;                       // 16B chunk 0..255
      async_copy_b128(kdst + 16u * j, ksrc + 16 * j);
    }
    const char* vsrc = Vg + (size_t)kb * 2;
    unsigned vdst = (unsigned)(uintptr_t)&Vs[buf][0][0];
#pragma unroll
    for (int c = 0; c < 2; ++c) {
      int j = tid + 128 * c;                       // 16B chunk 0..255
      int d = j >> 2, q = j & 3;
      async_copy_b128(vdst + (unsigned)(d * 64 + q * 16),
                      vsrc + (size_t)d * (Nlen * 2) + q * 16);
    }
  };

  stage_tile(0, 0);

  for (int it = 0; it < Nlen / 32; ++it) {
    const int cur = it & 1;
    const bool more = (it + 1) < (Nlen / 32);
    if (more) stage_tile((it + 1) * 32, cur ^ 1);
    // Release current tile while next tile's 4 async ops stay in flight.
    if (more) asm volatile("s_wait_asynccnt 4" ::: "memory");
    else      asm volatile("s_wait_asynccnt 0" ::: "memory");
    __syncthreads();

    v8f s0 = {}, s1 = {};
    const bf16_t* Kp = &Ks[cur][0][0];
    s0 = wmma_bf16(aq0, load_frag_b(Kp, HD), s0);
    s0 = wmma_bf16(aq1, load_frag_b(Kp + 32, HD), s0);
    s1 = wmma_bf16(aq0, load_frag_b(Kp + 16 * HD, HD), s1);
    s1 = wmma_bf16(aq1, load_frag_b(Kp + 16 * HD + 32, HD), s1);

    // Online softmax: row m = r + 8*half lives across the 16 lanes of a half.
#pragma unroll
    for (int r = 0; r < 8; ++r) {
      float e0 = s0[r] * scale, e1 = s1[r] * scale;
      float mx = fmaxf(e0, e1);
#pragma unroll
      for (int msk = 1; msk < 16; msk <<= 1)
        mx = fmaxf(mx, __shfl_xor(mx, msk, 32));
      float nm = fmaxf(mrow[r], mx);
      float p0 = __expf(e0 - nm), p1 = __expf(e1 - nm);
      float rs = p0 + p1;
#pragma unroll
      for (int msk = 1; msk < 16; msk <<= 1)
        rs += __shfl_xor(rs, msk, 32);
      float f = __expf(mrow[r] - nm);
      lrow[r] = lrow[r] * f + rs;
      mrow[r] = nm;
#pragma unroll
      for (int n = 0; n < 4; ++n) o[n][r] *= f;
      Pb[w][r + 8 * half][col] = (bf16_t)p0;
      Pb[w][r + 8 * half][col + 16] = (bf16_t)p1;
    }

    // O += P @ V ; P re-read from LDS in A-layout (wave-local ds ops).
    v16bf ap = load_frag_a(&Pb[w][0][0], 32);
#pragma unroll
    for (int n = 0; n < 4; ++n)
      o[n] = wmma_bf16(ap, load_frag_b(&Vs[cur][n * 16][0], 32), o[n]);

    __syncthreads();   // protect buffers from next iteration's async issue
  }

  // Normalize and write O in [b, t, h*64+d] (row-major bf16 for proj GEMM).
#pragma unroll
  for (int n = 0; n < 4; ++n)
#pragma unroll
    for (int r = 0; r < 8; ++r) {
      int t = m0 + r + 8 * half;
      int c = h * HD + n * 16 + col;
      Ob[((size_t)bidx * Nlen + t) * Cdim + c] = (bf16_t)(o[n][r] / lrow[r]);
    }
}

// ---------------- Output projection GEMM + bias (fp32 out) ----------------
__global__ void __launch_bounds__(32)
proj_kernel(const bf16_t* __restrict__ Xo, const bf16_t* __restrict__ W,
            const float* __restrict__ bias, float* __restrict__ out) {
  const int m0 = blockIdx.x * 64;
  const int n0 = blockIdx.y * 64;
  const int lane = threadIdx.x & 31;
  const int col = lane & 15, half = lane >> 4;

  v8f acc[4][4];
#pragma unroll
  for (int i = 0; i < 4; ++i)
#pragma unroll
    for (int j = 0; j < 4; ++j) { v8f z = {}; acc[i][j] = z; }

  for (int k0 = 0; k0 < Cdim; k0 += 32) {
    if (k0 + 32 < Cdim) {
      __builtin_prefetch(Xo + (m0 + 2 * lane) * Cdim + k0 + 32, 0, 1);
      __builtin_prefetch(W + (n0 + 2 * lane) * Cdim + k0 + 32, 0, 1);
    }
    v16bf a[4], b[4];
#pragma unroll
    for (int i = 0; i < 4; ++i) a[i] = load_frag_a(Xo + (m0 + 16 * i) * Cdim + k0, Cdim);
#pragma unroll
    for (int j = 0; j < 4; ++j) b[j] = load_frag_b(W + (n0 + 16 * j) * Cdim + k0, Cdim);
#pragma unroll
    for (int i = 0; i < 4; ++i)
#pragma unroll
      for (int j = 0; j < 4; ++j)
        acc[i][j] = wmma_bf16(a[i], b[j], acc[i][j]);
  }

#pragma unroll
  for (int j = 0; j < 4; ++j) {
    const int ncol = n0 + 16 * j + col;
    const float bj = bias[ncol];
#pragma unroll
    for (int i = 0; i < 4; ++i)
#pragma unroll
      for (int r = 0; r < 8; ++r) {
        int m = m0 + 16 * i + r + 8 * half;
        out[(size_t)m * Cdim + ncol] = acc[i][j][r] + bj;
      }
  }
}

// ---------------- host launch ----------------
extern "C" void kernel_launch(void* const* d_in, const int* in_sizes, int n_in,
                              void* d_out, int out_size, void* d_ws, size_t ws_size,
                              hipStream_t stream) {
  const float* x      = (const float*)d_in[0];
  const float* sinp   = (const float*)d_in[1];
  const float* cosp   = (const float*)d_in[2];
  const float* qkv_w  = (const float*)d_in[3];
  const float* proj_w = (const float*)d_in[4];
  const float* proj_b = (const float*)d_in[5];
  float* out = (float*)d_out;

  char* ws = (char*)d_ws;
  size_t off = 0;
  bf16_t* xbf   = (bf16_t*)(ws + off); off += (size_t)Mrows * Cdim * 2;        // 8 MB
  bf16_t* wqkv  = (bf16_t*)(ws + off); off += (size_t)TC * Cdim * 2;           // 6 MB
  bf16_t* wproj = (bf16_t*)(ws + off); off += (size_t)Cdim * Cdim * 2;         // 2 MB
  bf16_t* Qb    = (bf16_t*)(ws + off); off += (size_t)Bsz * Hn * Nlen * HD * 2; // 8 MB
  bf16_t* Kb    = (bf16_t*)(ws + off); off += (size_t)Bsz * Hn * Nlen * HD * 2; // 8 MB
  bf16_t* Vt    = (bf16_t*)(ws + off); off += (size_t)Bsz * Hn * HD * Nlen * 2; // 8 MB
  bf16_t* Ob    = (bf16_t*)(ws + off); off += (size_t)Bsz * Nlen * Cdim * 2;    // 8 MB

  int n4;
  n4 = Mrows * Cdim / 4;
  cvt_f32_bf16<<<(n4 + 255) / 256, 256, 0, stream>>>(x, xbf, n4);
  n4 = TC * Cdim / 4;
  cvt_f32_bf16<<<(n4 + 255) / 256, 256, 0, stream>>>(qkv_w, wqkv, n4);
  n4 = Cdim * Cdim / 4;
  cvt_f32_bf16<<<(n4 + 255) / 256, 256, 0, stream>>>(proj_w, wproj, n4);

  dim3 g1(Mrows / 64, TC / 64);
  qkv_rope_kernel<<<g1, 32, 0, stream>>>(xbf, wqkv, sinp, cosp, Qb, Kb, Vt);

  dim3 g2(Bsz * Hn, Nlen / 64);
  attn_kernel<<<g2, 128, 0, stream>>>(Qb, Kb, Vt, Ob);

  dim3 g3(Mrows / 64, Cdim / 64);
  proj_kernel<<<g3, 32, 0, stream>>>(Ob, wproj, proj_b, out);
}
